// GCNClassifierWithMetrics_2680059592847
// MI455X (gfx1250) — matrix operand order repaired and checked by reading, not compiled, not run on hardware
//
#include <hip/hip_runtime.h>
#include <hip/hip_bf16.h>

typedef __attribute__((ext_vector_type(16))) _Float16 v16h;
typedef __attribute__((ext_vector_type(8)))  _Float16 v8h;
typedef __attribute__((ext_vector_type(8)))  float    v8f;

__device__ __forceinline__ float silu_f(float t) { return t / (1.0f + __expf(-t)); }

// ---- WMMA fragment loaders (contiguous k order, same for A and B) ----
// A 16x32 f16: lane L -> row (L&15); 16 contiguous k-values starting at 16*(L>>4)
__device__ __forceinline__ v16h load_frag_a(const _Float16* __restrict__ base, int ld,
                                            int row_base, int kbase, int lane) {
    int m = lane & 15, grp = lane >> 4;
    const _Float16* p = base + (size_t)(row_base + m) * ld + kbase + grp * 16;
    v8h lo = *(const v8h*)p;
    v8h hi = *(const v8h*)(p + 8);
    v16h a;
#pragma unroll
    for (int i = 0; i < 8; ++i) { a[i] = lo[i]; a[i + 8] = hi[i]; }
    return a;
}
// B 32x16 f16 from a pre-transposed weight Wt[ncols][K]:
// lane L -> col (L&15); 16 contiguous k-values starting at 16*(L>>4)
__device__ __forceinline__ v16h load_frag_b(const _Float16* __restrict__ wt, int K,
                                            int col_base, int kbase, int lane) {
    int n = lane & 15, grp = lane >> 4;
    return *(const v16h*)(wt + (size_t)(col_base + n) * K + kbase + grp * 16);
}

// ---- GEMM1: hlin = x@W1 ; r1 = silu(x@Wr1 + br1). x:[N,128] f16, W*t:[64,128] f16 ----
__global__ void k_gemm1(const _Float16* __restrict__ xh,
                        const _Float16* __restrict__ w1t,
                        const _Float16* __restrict__ wr1t,
                        const float* __restrict__ br1,
                        float* __restrict__ hlin, float* __restrict__ r1, int n) {
    int wave = (int)((blockIdx.x * blockDim.x + threadIdx.x) >> 5);
    int lane = threadIdx.x & 31;
    int row_base = wave * 16;
    if (row_base >= n) return;              // wave-uniform: EXEC stays all-ones
    v8f zero = {0.f,0.f,0.f,0.f,0.f,0.f,0.f,0.f};
    v8f accW[4], accR[4];
#pragma unroll
    for (int t = 0; t < 4; ++t) { accW[t] = zero; accR[t] = zero; }
#pragma unroll
    for (int kc = 0; kc < 4; ++kc) {
        v16h a = load_frag_a(xh, 128, row_base, kc * 32, lane);
#pragma unroll
        for (int nt = 0; nt < 4; ++nt) {
            v16h bW = load_frag_b(w1t, 128, nt * 16, kc * 32, lane);
            accW[nt] = __builtin_amdgcn_wmma_f32_16x16x32_f16(false, a, false, bW,
                                                              (short)0, accW[nt], false, false);
            v16h bR = load_frag_b(wr1t, 128, nt * 16, kc * 32, lane);
            accR[nt] = __builtin_amdgcn_wmma_f32_16x16x32_f16(false, a, false, bR,
                                                              (short)0, accR[nt], false, false);
        }
    }
    int n0 = lane & 15, grp = lane >> 4;
#pragma unroll
    for (int nt = 0; nt < 4; ++nt) {
        int col = nt * 16 + n0;
        float bias = br1[col];
#pragma unroll
        for (int v = 0; v < 8; ++v) {
            int row = row_base + v + 8 * grp;
            hlin[(size_t)row * 64 + col] = accW[nt][v];
            r1[(size_t)row * 64 + col] = silu_f(accR[nt][v] + bias);
        }
    }
}

// ---- GEMM2: zlin = h@W2 ; r2 = silu(h@Wr2 + br2). h:[N,64] f16, W*t:[16,64] f16 ----
__global__ void k_gemm2(const _Float16* __restrict__ hh,
                        const _Float16* __restrict__ w2t,
                        const _Float16* __restrict__ wr2t,
                        const float* __restrict__ br2,
                        float* __restrict__ zlin, float* __restrict__ r2, int n) {
    int wave = (int)((blockIdx.x * blockDim.x + threadIdx.x) >> 5);
    int lane = threadIdx.x & 31;
    int row_base = wave * 16;
    if (row_base >= n) return;
    v8f accZ = {0.f,0.f,0.f,0.f,0.f,0.f,0.f,0.f};
    v8f accR = accZ;
#pragma unroll
    for (int kc = 0; kc < 2; ++kc) {
        v16h a  = load_frag_a(hh, 64, row_base, kc * 32, lane);
        v16h bZ = load_frag_b(w2t, 64, 0, kc * 32, lane);
        accZ = __builtin_amdgcn_wmma_f32_16x16x32_f16(false, a, false, bZ, (short)0, accZ, false, false);
        v16h bR = load_frag_b(wr2t, 64, 0, kc * 32, lane);
        accR = __builtin_amdgcn_wmma_f32_16x16x32_f16(false, a, false, bR, (short)0, accR, false, false);
    }
    int n0 = lane & 15, grp = lane >> 4;
    float bias = br2[n0];
#pragma unroll
    for (int v = 0; v < 8; ++v) {
        int row = row_base + v + 8 * grp;
        zlin[(size_t)row * 16 + n0] = accZ[v];
        r2[(size_t)row * 16 + n0] = silu_f(accR[v] + bias);
    }
}

// ---- helpers ----
__global__ void k_f32_to_f16(const float* __restrict__ s, _Float16* __restrict__ d, int total) {
    int i = blockIdx.x * blockDim.x + threadIdx.x;
    if (i < total) d[i] = (_Float16)s[i];
}
// W[K][Ncol] (row-major) -> Wt[Ncol][K] f16
__global__ void k_transpose_h(const float* __restrict__ w, _Float16* __restrict__ wt, int K, int Ncol) {
    int i = blockIdx.x * blockDim.x + threadIdx.x;
    if (i >= K * Ncol) return;
    int ncol = i / K, k = i % K;
    wt[i] = (_Float16)w[(size_t)k * Ncol + ncol];
}
__global__ void k_fill(float* __restrict__ p, float v, int total) {
    int i = blockIdx.x * blockDim.x + threadIdx.x;
    if (i < total) p[i] = v;
}
__global__ void k_degree(const int* __restrict__ dst, float* __restrict__ deg, int E) {
    int i = blockIdx.x * blockDim.x + threadIdx.x;
    if (i < E) atomicAdd(&deg[dst[i]], 1.0f);
}
__global__ void k_rsqrt(const float* __restrict__ deg, float* __restrict__ dinv, int n) {
    int i = blockIdx.x * blockDim.x + threadIdx.x;
    if (i < n) { float d = deg[i]; dinv[i] = d > 0.0f ? rsqrtf(fmaxf(d, 1.0f)) : 0.0f; }
}
// agg init with self-loop term: agg[i,c] = lin[i,c] * dinv[i]^2
__global__ void k_selfloop(const float* __restrict__ lin, const float* __restrict__ dinv,
                           float* __restrict__ agg, int n, int C) {
    int i = blockIdx.x * blockDim.x + threadIdx.x;
    if (i >= n * C) return;
    float di = dinv[i / C];
    agg[i] = lin[i] * di * di;
}
// scatter-add over edges: agg[dst] += lin[src] * dinv[src]*dinv[dst], 4 channels/thread
__global__ void k_edge_agg(const int* __restrict__ src, const int* __restrict__ dst,
                           const float* __restrict__ dinv, const float* __restrict__ lin,
                           float* __restrict__ agg, int E, int C) {
    int tid = blockIdx.x * blockDim.x + threadIdx.x;
    int per = C >> 2;
    int e = tid / per;
    if (e >= E) return;
    int c = (tid % per) << 2;
    int s = src[e], d = dst[e];
    float w = dinv[s] * dinv[d];
    float4 v = *(const float4*)(lin + (size_t)s * C + c);
    float* o = agg + (size_t)d * C + c;
    atomicAdd(o + 0, v.x * w);
    atomicAdd(o + 1, v.y * w);
    atomicAdd(o + 2, v.z * w);
    atomicAdd(o + 3, v.w * w);
}
// h = silu(agg1 + b1) + alpha1*r1  -> f16 for next GEMM
__global__ void k_make_h(const float* __restrict__ agg1, const float* __restrict__ b1,
                         const float* __restrict__ r1, const float* __restrict__ alpha1,
                         _Float16* __restrict__ hh, int n) {
    int i = blockIdx.x * blockDim.x + threadIdx.x;
    if (i >= n * 64) return;
    float t = agg1[i] + b1[i & 63];
    hh[i] = (_Float16)(silu_f(t) + alpha1[0] * r1[i]);
}
// z = agg2 + b2 + alpha2*r2, atomic pool into per-graph sums
__global__ void k_pool(const float* __restrict__ agg2, const float* __restrict__ b2,
                       const float* __restrict__ r2, const float* __restrict__ alpha2,
                       const int* __restrict__ batch, float* __restrict__ gsum, int n) {
    int i = blockIdx.x * blockDim.x + threadIdx.x;
    if (i >= n * 16) return;
    int node = i >> 4, c = i & 15;
    float z = agg2[i] + b2[c] + alpha2[0] * r2[i];
    atomicAdd(&gsum[(size_t)batch[node] * 16 + c], z);
}
__global__ void k_count(const int* __restrict__ batch, float* __restrict__ gcnt, int n) {
    int i = blockIdx.x * blockDim.x + threadIdx.x;
    if (i < n) atomicAdd(&gcnt[batch[i]], 1.0f);
}
// 4 tiny MLPs (1->8->16), output laid out metric-major [4*16]
__global__ void k_metrics(const float* __restrict__ tol, const float* __restrict__ cost,
                          const float* __restrict__ time_, const float* __restrict__ qty,
                          const float* __restrict__ mW1, const float* __restrict__ mb1,
                          const float* __restrict__ mW2, const float* __restrict__ mb2,
                          float* __restrict__ metrics) {
    int t = threadIdx.x;
    if (t >= 64) return;
    int m = t >> 4, c = t & 15;
    float s = (m == 0) ? tol[0] : (m == 1) ? cost[0] : (m == 2) ? time_[0] : qty[0];
    float acc = mb2[m * 16 + c];
#pragma unroll
    for (int j = 0; j < 8; ++j) {
        float hj = silu_f(s * mW1[m * 8 + j] + mb1[m * 8 + j]);
        acc += hj * mW2[m * 128 + j * 16 + c];
    }
    metrics[t] = acc;
}
__global__ void k_head1(const float* __restrict__ gsum, const float* __restrict__ gcnt,
                        const float* __restrict__ metrics, const float* __restrict__ Wf1,
                        const float* __restrict__ bf1, float* __restrict__ hidden, int G) {
    int tid = blockIdx.x * blockDim.x + threadIdx.x;
    if (tid >= G * 80) return;
    int g = tid / 80, j = tid % 80;
    float inv = 1.0f / fmaxf(gcnt[g], 1.0f);
    float acc = bf1[j];
#pragma unroll
    for (int k = 0; k < 16; ++k) acc += (gsum[g * 16 + k] * inv) * Wf1[k * 80 + j];
    for (int k = 0; k < 64; ++k) acc += metrics[k] * Wf1[(16 + k) * 80 + j];
    hidden[tid] = silu_f(acc);
}
__global__ void k_head2(const float* __restrict__ hidden, const float* __restrict__ Wf2,
                        const float* __restrict__ bf2, float* __restrict__ out, int G) {
    int tid = blockIdx.x * blockDim.x + threadIdx.x;
    if (tid >= G * 10) return;
    int g = tid / 10, c = tid % 10;
    float acc = bf2[c];
#pragma unroll
    for (int k = 0; k < 80; ++k) acc += hidden[g * 80 + k] * Wf2[k * 10 + c];
    out[tid] = acc;
}

static inline int cdiv(long long a, long long b) { return (int)((a + b - 1) / b); }

extern "C" void kernel_launch(void* const* d_in, const int* in_sizes, int n_in,
                              void* d_out, int out_size, void* d_ws, size_t ws_size,
                              hipStream_t stream) {
    const float* x      = (const float*)d_in[0];
    const int*   ei     = (const int*)d_in[1];
    const int*   batch  = (const int*)d_in[2];
    const float* tol    = (const float*)d_in[3];
    const float* cost   = (const float*)d_in[4];
    const float* time_  = (const float*)d_in[5];
    const float* qty    = (const float*)d_in[6];
    const float* W1     = (const float*)d_in[7];
    const float* b1     = (const float*)d_in[8];
    const float* W2     = (const float*)d_in[9];
    const float* b2     = (const float*)d_in[10];
    const float* Wr1    = (const float*)d_in[11];
    const float* br1    = (const float*)d_in[12];
    const float* Wr2    = (const float*)d_in[13];
    const float* br2    = (const float*)d_in[14];
    const float* alpha1 = (const float*)d_in[15];
    const float* alpha2 = (const float*)d_in[16];
    const float* mW1    = (const float*)d_in[17];
    const float* mb1    = (const float*)d_in[18];
    const float* mW2    = (const float*)d_in[19];
    const float* mb2    = (const float*)d_in[20];
    const float* Wf1    = (const float*)d_in[21];
    const float* bf1    = (const float*)d_in[22];
    const float* Wf2    = (const float*)d_in[23];
    const float* bf2    = (const float*)d_in[24];

    const int N = in_sizes[0] / 128;
    const int E = in_sizes[1] / 2;
    const int G = out_size / 10;
    const int* src = ei;
    const int* dst = ei + E;

    // ---- carve workspace (256B-aligned slabs) ----
    char* ws = (char*)d_ws;
    size_t off = 0;
    auto alloc = [&](size_t bytes) -> void* {
        void* p = ws + off;
        off += (bytes + 255) & ~(size_t)255;
        return p;
    };
    _Float16* xh    = (_Float16*)alloc((size_t)N * 128 * 2);
    _Float16* w1t   = (_Float16*)alloc(64 * 128 * 2);
    _Float16* wr1t  = (_Float16*)alloc(64 * 128 * 2);
    _Float16* w2t   = (_Float16*)alloc(16 * 64 * 2);
    _Float16* wr2t  = (_Float16*)alloc(16 * 64 * 2);
    float*    deg   = (float*)alloc((size_t)N * 4);
    float*    dinv  = (float*)alloc((size_t)N * 4);
    float*    hlin  = (float*)alloc((size_t)N * 64 * 4);
    float*    r1    = (float*)alloc((size_t)N * 64 * 4);
    float*    agg1  = (float*)alloc((size_t)N * 64 * 4);
    _Float16* hh    = (_Float16*)alloc((size_t)N * 64 * 2);
    float*    zlin  = (float*)alloc((size_t)N * 16 * 4);
    float*    r2    = (float*)alloc((size_t)N * 16 * 4);
    float*    agg2  = (float*)alloc((size_t)N * 16 * 4);
    float*    gsum  = (float*)alloc((size_t)G * 16 * 4);
    float*    gcnt  = (float*)alloc((size_t)G * 4);
    float*    metrics = (float*)alloc(64 * 4);
    float*    hidden  = (float*)alloc((size_t)G * 80 * 4);
    (void)ws_size;

    const int TB = 256;

    // precision conversion + weight transposes
    k_f32_to_f16<<<cdiv((long long)N * 128, TB), TB, 0, stream>>>(x, xh, N * 128);
    k_transpose_h<<<cdiv(128 * 64, TB), TB, 0, stream>>>(W1,  w1t,  128, 64);
    k_transpose_h<<<cdiv(128 * 64, TB), TB, 0, stream>>>(Wr1, wr1t, 128, 64);
    k_transpose_h<<<cdiv(64 * 16, TB), TB, 0, stream>>>(W2,  w2t,  64, 16);
    k_transpose_h<<<cdiv(64 * 16, TB), TB, 0, stream>>>(Wr2, wr2t, 64, 16);

    // symmetric GCN normalization (self-loop folded into deg init)
    k_fill<<<cdiv(N, TB), TB, 0, stream>>>(deg, 1.0f, N);
    k_degree<<<cdiv(E, TB), TB, 0, stream>>>(dst, deg, E);
    k_rsqrt<<<cdiv(N, TB), TB, 0, stream>>>(deg, dinv, N);

    // layer 1: WMMA GEMMs, then edge aggregation
    {
        int waves = cdiv(N, 16);
        k_gemm1<<<cdiv((long long)waves * 32, TB), TB, 0, stream>>>(xh, w1t, wr1t, br1, hlin, r1, N);
    }
    k_selfloop<<<cdiv((long long)N * 64, TB), TB, 0, stream>>>(hlin, dinv, agg1, N, 64);
    k_edge_agg<<<cdiv((long long)E * 16, TB), TB, 0, stream>>>(src, dst, dinv, hlin, agg1, E, 64);
    k_make_h<<<cdiv((long long)N * 64, TB), TB, 0, stream>>>(agg1, b1, r1, alpha1, hh, N);

    // layer 2
    {
        int waves = cdiv(N, 16);
        k_gemm2<<<cdiv((long long)waves * 32, TB), TB, 0, stream>>>(hh, w2t, wr2t, br2, zlin, r2, N);
    }
    k_selfloop<<<cdiv((long long)N * 16, TB), TB, 0, stream>>>(zlin, dinv, agg2, N, 16);
    k_edge_agg<<<cdiv((long long)E * 4, TB), TB, 0, stream>>>(src, dst, dinv, zlin, agg2, E, 16);

    // mean-pool per graph
    k_fill<<<cdiv(G * 16, TB), TB, 0, stream>>>(gsum, 0.0f, G * 16);
    k_fill<<<cdiv(G, TB), TB, 0, stream>>>(gcnt, 0.0f, G);
    k_pool<<<cdiv((long long)N * 16, TB), TB, 0, stream>>>(agg2, b2, r2, alpha2, batch, gsum, N);
    k_count<<<cdiv(N, TB), TB, 0, stream>>>(batch, gcnt, N);

    // metrics embedding + classifier head
    k_metrics<<<1, 64, 0, stream>>>(tol, cost, time_, qty, mW1, mb1, mW2, mb2, metrics);
    k_head1<<<cdiv(G * 80, TB), TB, 0, stream>>>(gsum, gcnt, metrics, Wf1, bf1, hidden, G);
    k_head2<<<cdiv(G * 10, TB), TB, 0, stream>>>(hidden, Wf2, bf2, (float*)d_out, G);
}
